// joint_GRU_75531294867991
// MI455X (gfx1250) — compile-verified, matrix operand order checked by reading, not run on hardware
//
#include <hip/hip_runtime.h>
#include <hip/hip_bf16.h>

// Fused GRU-style cell for MI455X (gfx1250, wave32, WMMA bf16 w/ f32 acc).
//
// B = 65536 rows, K = H = 256.  Two kernels:
//   1) pack_weights: f32 -> bf16, pre-swizzled into WMMA B-fragment order
//      (per 32x16 K-N tile: 32 lanes x 16 bf16 contiguous = 1 KB/tile) in d_ws.
//      Needs 6 * 256*256 * 2 = 768 KB of workspace.
//   2) gru_fused: per block 128 rows x 256 cols; each wave owns ONE 16-col
//      N-tile across all eight 16-row M-tiles, so every B fragment loaded from
//      L2 is reused by 8 WMMAs (A fragments stream from LDS).  The mgn LDS
//      buffer is dead after stage 1 and is reused for g = joint*r.

typedef __attribute__((ext_vector_type(16))) __bf16 v16bf;
typedef __attribute__((ext_vector_type(8)))  float  v8f;

#define ROWS 128         // rows per workgroup
#define MTILES (ROWS / 16)
#define HDIM 256         // hidden size == K
#define PITCH 264        // LDS row pitch in bf16 elems (528 B) -> conflict-free b128
#define MAT_ELEMS 65536  // 256*256 elems per packed matrix

// ---------------- weight pack kernel ----------------
// Packed tile (kt, nt) of a [K=256 x N=256] row-major f32 matrix:
//   lane l (0..31), elem i (0..15):
//     n  = nt*16 + (l & 15)
//     kk = (i < 8 ? i : 16 + (i - 8)) + (l < 16 ? 0 : 8)
//     k  = kt*32 + kk
// (mirrors the 16-bit A-fragment layout with M<->N roles swapped)
__device__ __forceinline__ unsigned short f32_to_bf16_bits(float x) {
    union { float f; unsigned u; } a; a.f = x;
    unsigned u = a.u;
    u += 0x7fffu + ((u >> 16) & 1u);   // round to nearest even
    return (unsigned short)(u >> 16);
}

__global__ __launch_bounds__(256) void pack_weights(
    const float* __restrict__ Wr, const float* __restrict__ Wz,
    const float* __restrict__ Wh, const float* __restrict__ Ur,
    const float* __restrict__ Uz, const float* __restrict__ Uh,
    unsigned short* __restrict__ dst)
{
    int idx = blockIdx.x * 256 + threadIdx.x;     // 0 .. 6*65536-1
    int mat = idx >> 16;
    int rem = idx & 65535;
    int tile = rem >> 9;          // 512 elems per tile
    int lane = (rem >> 4) & 31;
    int i    = rem & 15;
    int kt = tile >> 4;
    int nt = tile & 15;

    int n  = nt * 16 + (lane & 15);
    int kk = ((i < 8) ? i : (16 + (i - 8))) + ((lane < 16) ? 0 : 8);
    int k  = kt * 32 + kk;

    const float* src = (mat == 0) ? Wr : (mat == 1) ? Wz : (mat == 2) ? Wh
                     : (mat == 3) ? Ur : (mat == 4) ? Uz : Uh;
    dst[idx] = f32_to_bf16_bits(src[k * HDIM + n]);
}

// ---------------- fragment loaders ----------------
__device__ __forceinline__ v16bf load_a_frag(const __bf16* sm, int row0, int k0, int lane) {
    // 16-bit A 16x32: lanes 0-15 -> M=lane, K {k0..k0+7, k0+16..k0+23}
    //                 lanes16-31 -> M=lane-16, K {k0+8..k0+15, k0+24..k0+31}
    int half = lane >> 4;
    int r    = lane & 15;
    const __bf16* base = sm + (row0 + r) * PITCH + k0 + (half << 3);
    union { uint4 q[2]; v16bf v; } u;
    u.q[0] = *reinterpret_cast<const uint4*>(base);        // 8 bf16
    u.q[1] = *reinterpret_cast<const uint4*>(base + 16);   // 8 bf16
    return u.v;
}

__device__ __forceinline__ v16bf load_b_frag(const __bf16* wm, int tileIdx, int lane) {
    const __bf16* p = wm + tileIdx * 512 + lane * 16;      // 32 B per lane, coalesced
    union { uint4 q[2]; v16bf v; } u;
    u.q[0] = *reinterpret_cast<const uint4*>(p);
    u.q[1] = *reinterpret_cast<const uint4*>(p + 8);
    return u.v;
}

__device__ __forceinline__ v8f wmma_bf16(v16bf a, v16bf b, v8f c) {
    return __builtin_amdgcn_wmma_f32_16x16x32_bf16(false, a, false, b,
                                                   (short)0, c, false, false);
}

__device__ __forceinline__ float sigmoid_f(float x) {
    return 1.0f / (1.0f + __expf(-x));
}

// ---------------- fused GRU kernel ----------------
__global__ __launch_bounds__(512) void gru_fused(
    const float* __restrict__ joint, const float* __restrict__ mgn,
    const unsigned short* __restrict__ wpk_,
    const float* __restrict__ br, const float* __restrict__ bz,
    const float* __restrict__ bh, float* __restrict__ out)
{
    const __bf16* wpk = reinterpret_cast<const __bf16*>(wpk_);

    // s_x holds mgn during stage 1, then is recycled as g = joint*r for stage 2
    __shared__ __bf16 s_x[ROWS * PITCH];
    __shared__ __bf16 s_joint[ROWS * PITCH];

    const int tid = threadIdx.x;
    const int b0  = blockIdx.x * ROWS;

    // ---- stage activations to LDS as bf16 (one HBM read each) ----
    #pragma unroll 4
    for (int e = 0; e < (ROWS * HDIM) / 512; ++e) {
        int idx = e * 512 + tid;
        int r = idx >> 8, c = idx & 255;
        s_x[r * PITCH + c]     = (__bf16)mgn[(size_t)(b0 + r) * HDIM + c];
        s_joint[r * PITCH + c] = (__bf16)joint[(size_t)(b0 + r) * HDIM + c];
    }
    __syncthreads();

    const int nt   = tid >> 5;         // wave id == N-tile (0..15)
    const int lane = tid & 31;
    const int nlow = lane & 15;
    const int hof  = (lane >> 4) ? 8 : 0;
    const int n    = nt * 16 + nlow;   // this thread's output column

    v8f zero8 = {0.f, 0.f, 0.f, 0.f, 0.f, 0.f, 0.f, 0.f};
    v8f accR[MTILES], accZ[MTILES], accH[MTILES];   // indexed by M-tile
    #pragma unroll
    for (int mt = 0; mt < MTILES; ++mt) {
        accR[mt] = zero8; accZ[mt] = zero8; accH[mt] = zero8;
    }

    // ---- stage 1: pre_r = mgn@Wr + joint@Ur ; pre_z = mgn@Wz + joint@Uz ;
    //               x_h = mgn@Wh.  One B fragment per matrix per k-step,
    //               reused across all 8 M-tiles. ----
    for (int k0 = 0; k0 < HDIM; k0 += 32) {
        int tIdx = (k0 >> 5) * 16 + nt;
        v16bf Bwr = load_b_frag(wpk + 0 * MAT_ELEMS, tIdx, lane);
        v16bf Bwz = load_b_frag(wpk + 1 * MAT_ELEMS, tIdx, lane);
        v16bf Bwh = load_b_frag(wpk + 2 * MAT_ELEMS, tIdx, lane);
        v16bf Bur = load_b_frag(wpk + 3 * MAT_ELEMS, tIdx, lane);
        v16bf Buz = load_b_frag(wpk + 4 * MAT_ELEMS, tIdx, lane);
        #pragma unroll
        for (int mt = 0; mt < MTILES; ++mt) {
            v16bf Am = load_a_frag(s_x,     mt * 16, k0, lane);
            v16bf Aj = load_a_frag(s_joint, mt * 16, k0, lane);
            accR[mt] = wmma_bf16(Am, Bwr, accR[mt]);
            accR[mt] = wmma_bf16(Aj, Bur, accR[mt]);
            accZ[mt] = wmma_bf16(Am, Bwz, accZ[mt]);
            accZ[mt] = wmma_bf16(Aj, Buz, accZ[mt]);
            accH[mt] = wmma_bf16(Am, Bwh, accH[mt]);
        }
    }

    // all stage-1 reads of s_x (mgn) must retire before we overwrite it with g
    __syncthreads();

    // ---- r = sigmoid(pre_r + b_r);  g = joint * r  -> LDS (reusing s_x) ----
    // C/D layout: VGPR v -> M = v (lanes 0-15) or v+8 (lanes 16-31); N = lane&15
    const float brv = br[n];
    #pragma unroll
    for (int mt = 0; mt < MTILES; ++mt) {
        #pragma unroll
        for (int v = 0; v < 8; ++v) {
            int m = mt * 16 + v + hof;
            float rv = sigmoid_f(accR[mt][v] + brv);
            float jv = (float)s_joint[m * PITCH + n];
            s_x[m * PITCH + n] = (__bf16)(rv * jv);
        }
    }
    __syncthreads();

    // ---- stage 2: h_pre = x_h + g @ U_h ----
    for (int k0 = 0; k0 < HDIM; k0 += 32) {
        int tIdx = (k0 >> 5) * 16 + nt;
        v16bf Buh = load_b_frag(wpk + 5 * MAT_ELEMS, tIdx, lane);
        #pragma unroll
        for (int mt = 0; mt < MTILES; ++mt) {
            v16bf Ag = load_a_frag(s_x, mt * 16, k0, lane);
            accH[mt] = wmma_bf16(Ag, Buh, accH[mt]);
        }
    }

    // ---- epilogue: h~ = tanh(h_pre + b_h); z = sigmoid(pre_z + b_z);
    //      out = h~ + z * (joint - h~) ----
    const float bzv = bz[n];
    const float bhv = bh[n];
    #pragma unroll
    for (int mt = 0; mt < MTILES; ++mt) {
        #pragma unroll
        for (int v = 0; v < 8; ++v) {
            int m = mt * 16 + v + hof;
            float zv = sigmoid_f(accZ[mt][v] + bzv);
            float ht = tanhf(accH[mt][v] + bhv);
            float jt = joint[(size_t)(b0 + m) * HDIM + n];
            out[(size_t)(b0 + m) * HDIM + n] = ht + zv * (jt - ht);
        }
    }
}

// ---------------- host launch ----------------
extern "C" void kernel_launch(void* const* d_in, const int* in_sizes, int n_in,
                              void* d_out, int out_size, void* d_ws, size_t ws_size,
                              hipStream_t stream) {
    const float* joint = (const float*)d_in[0];
    const float* mgn   = (const float*)d_in[1];
    const float* Wr    = (const float*)d_in[2];
    const float* Wz    = (const float*)d_in[3];
    const float* Wh    = (const float*)d_in[4];
    const float* Ur    = (const float*)d_in[5];
    const float* Uz    = (const float*)d_in[6];
    const float* Uh    = (const float*)d_in[7];
    const float* br    = (const float*)d_in[8];
    const float* bz    = (const float*)d_in[9];
    const float* bh    = (const float*)d_in[10];
    float* outp        = (float*)d_out;
    unsigned short* wpk = (unsigned short*)d_ws;   // needs 768 KB

    int Bn = in_sizes[0] / HDIM;                   // 65536

    // 6 matrices * 65536 elems, 256 threads/block
    pack_weights<<<(6 * MAT_ELEMS) / 256, 256, 0, stream>>>(Wr, Wz, Wh, Ur, Uz, Uh, wpk);
    gru_fused<<<Bn / ROWS, 512, 0, stream>>>(joint, mgn, wpk, br, bz, bh, outp);
}
